// TFMModel_76398878261794
// MI455X (gfx1250) — matrix operand order repaired and checked by reading, not compiled, not run on hardware
//
#include <hip/hip_runtime.h>
#include <stdint.h>
#include <stddef.h>

// ---------------- model constants ----------------
#define BB     16
#define LIN    512
#define OUTSEQ 24
#define FIN    64
#define FOUT   32
#define HDIM   512
#define NH     8
#define HD     64
#define FFD    2048
#define NLAY   6

typedef __bf16 bf16_t;
typedef __attribute__((ext_vector_type(8)))  __bf16 v8bf;
typedef __attribute__((ext_vector_type(16))) __bf16 v16bf;
typedef __attribute__((ext_vector_type(8)))  float  v8f;

__device__ __forceinline__ bf16_t f2bf(float f) {
    unsigned u = __builtin_bit_cast(unsigned, f);
    unsigned r = (u + 0x7FFFu + ((u >> 16) & 1u)) >> 16;    // round-to-nearest-even
    unsigned short h = (unsigned short)r;
    return __builtin_bit_cast(bf16_t, h);
}
__device__ __forceinline__ float bf2f(bf16_t b) {
    unsigned short h = __builtin_bit_cast(unsigned short, b);
    unsigned u = ((unsigned)h) << 16;
    return __builtin_bit_cast(float, u);
}
__device__ __forceinline__ v16bf catbf(v8bf lo, v8bf hi) {
    return __builtin_shufflevector(lo, hi, 0,1,2,3,4,5,6,7,8,9,10,11,12,13,14,15);
}

// ---------------- batched GEMM: C = act(A(MxK) * W(NxK)^T + bias) ----------------
// A,W bf16; C written as f32 and/or bf16. One wave = 32x32 tile (2x2 WMMA),
// 8 waves/block = 128x64 block tile. K must be a multiple of 32.
__global__ __launch_bounds__(256) void k_gemm(
    const bf16_t* __restrict__ A, long long lda, long long sA,
    const bf16_t* __restrict__ W, long long ldw, long long sW,
    const float*  __restrict__ bias,
    float*  __restrict__ Cf, long long sCf,
    bf16_t* __restrict__ Cb, long long sCb,
    long long ldc, int M, int N, int K, int relu)
{
    const int z = blockIdx.z;
    A += (long long)z * sA;
    W += (long long)z * sW;
    if (Cf) Cf += (long long)z * sCf;
    if (Cb) Cb += (long long)z * sCb;

    const int wave = threadIdx.x >> 5;
    const int lane = threadIdx.x & 31;
    const int m0 = blockIdx.y * 128 + (wave >> 1) * 32;
    const int n0 = blockIdx.x * 64  + (wave & 1)  * 32;
    if (m0 >= M || n0 >= N) return;                 // wave-uniform exit

    const int lr = lane & 15;
    const int hi = lane >> 4;                       // 0 or 1
    const int kA = hi * 8;                          // A frag: K sub-offset (ISA 16-bit A layout)
    const int kB = hi * 8;                          // B frag: K sub-offset

    int ra0 = m0 + lr;       if (ra0 > M - 1) ra0 = M - 1;
    int ra1 = m0 + 16 + lr;  if (ra1 > M - 1) ra1 = M - 1;
    int rb0 = n0 + lr;       if (rb0 > N - 1) rb0 = N - 1;
    int rb1 = n0 + 16 + lr;  if (rb1 > N - 1) rb1 = N - 1;

    const bf16_t* pa0 = A + (long long)ra0 * lda + kA;
    const bf16_t* pa1 = A + (long long)ra1 * lda + kA;
    const bf16_t* pb0 = W + (long long)rb0 * ldw + kB;
    const bf16_t* pb1 = W + (long long)rb1 * ldw + kB;

    v8f zero8;
    for (int r = 0; r < 8; ++r) zero8[r] = 0.0f;
    v8f c00 = zero8, c01 = zero8, c10 = zero8, c11 = zero8;

    for (int k = 0; k < K; k += 32) {
        v16bf a0 = catbf(*(const v8bf*)(pa0 + k), *(const v8bf*)(pa0 + k + 16));
        v16bf a1 = catbf(*(const v8bf*)(pa1 + k), *(const v8bf*)(pa1 + k + 16));
        v16bf b0 = catbf(*(const v8bf*)(pb0 + k), *(const v8bf*)(pb0 + k + 16));
        v16bf b1 = catbf(*(const v8bf*)(pb1 + k), *(const v8bf*)(pb1 + k + 16));
        c00 = __builtin_amdgcn_wmma_f32_16x16x32_bf16(false, a0, false, b0, (short)0, c00, false, false);
        c01 = __builtin_amdgcn_wmma_f32_16x16x32_bf16(false, a0, false, b1, (short)0, c01, false, false);
        c10 = __builtin_amdgcn_wmma_f32_16x16x32_bf16(false, a1, false, b0, (short)0, c10, false, false);
        c11 = __builtin_amdgcn_wmma_f32_16x16x32_bf16(false, a1, false, b1, (short)0, c11, false, false);
    }

    // C layout: VGPR r, lanes 0-15 -> (M=r, N=lane); lanes 16-31 -> (M=r+8, N=lane-16)
    const int crow = m0 + hi * 8;
    const int ccol = n0 + lr;
    v8f accs[2][2] = { { c00, c01 }, { c10, c11 } };
    for (int i = 0; i < 2; ++i) {
        for (int j = 0; j < 2; ++j) {
            int col = ccol + j * 16;
            if (col >= N) continue;
            float bv = bias ? bias[col] : 0.0f;
            for (int r = 0; r < 8; ++r) {
                int row = crow + i * 16 + r;
                if (row >= M) continue;
                float v = accs[i][j][r] + bv;
                if (relu) v = fmaxf(v, 0.0f);
                long long off = (long long)row * ldc + col;
                if (Cf) Cf[off] = v;
                if (Cb) Cb[off] = f2bf(v);
            }
        }
    }
}

// ---------------- elementwise / epilogue kernels ----------------
__global__ __launch_bounds__(256) void k_cvt(const float* __restrict__ in,
                                             bf16_t* __restrict__ out, long long n) {
    long long i = (long long)blockIdx.x * 256 + threadIdx.x;
    if (i < n) out[i] = f2bf(in[i]);
}

__global__ __launch_bounds__(256) void k_zero(float* __restrict__ p, long long n) {
    long long i = (long long)blockIdx.x * 256 + threadIdx.x;
    if (i < n) p[i] = 0.0f;
}

// RoPE over rows of (rows, HDIM); pos = row % Ls (rows are batch-major). grid = rows, block = 256.
__global__ __launch_bounds__(256) void k_rope(const float* __restrict__ in,
                                              float* __restrict__ outf,
                                              bf16_t* __restrict__ outb, int Ls) {
    int row = blockIdx.x;
    int pr  = threadIdx.x;            // 0..255 : pair index within row
    int head = pr >> 5;               // /32
    int i    = pr & 31;
    int pos  = row % Ls;
    float inv = powf(10000.0f, -((float)(2 * i)) / 64.0f);
    float f = (float)pos * inv;
    float c = cosf(f), s = sinf(f);
    long long base = (long long)row * HDIM + head * HD + 2 * i;
    float e = in[base], o = in[base + 1];
    float re = e * c - o * s;
    float ro = e * s + o * c;
    outf[base] = re;  outf[base + 1] = ro;
    outb[base] = f2bf(re);  outb[base + 1] = f2bf(ro);
}

// LayerNorm over H: y = (x+res - mean)/sqrt(var+eps)*g + b. grid = rows.
__global__ __launch_bounds__(256) void k_ln(const float* __restrict__ x,
                                            const float* __restrict__ res,
                                            const float* __restrict__ g,
                                            const float* __restrict__ b,
                                            float* __restrict__ outf,
                                            bf16_t* __restrict__ outb, int H) {
    int row = blockIdx.x;
    const float* xr = x + (long long)row * H;
    const float* rr = res ? res + (long long)row * H : nullptr;
    __shared__ float s1[256], s2[256];
    float s = 0.f, ss = 0.f;
    for (int i = threadIdx.x; i < H; i += 256) {
        float v = xr[i] + (rr ? rr[i] : 0.f);
        s += v; ss += v * v;
    }
    s1[threadIdx.x] = s; s2[threadIdx.x] = ss;
    __syncthreads();
    for (int o = 128; o > 0; o >>= 1) {
        if (threadIdx.x < o) { s1[threadIdx.x] += s1[threadIdx.x + o]; s2[threadIdx.x] += s2[threadIdx.x + o]; }
        __syncthreads();
    }
    float mean = s1[0] / H;
    float var  = s2[0] / H - mean * mean;
    float rinv = rsqrtf(var + 1e-5f);
    for (int i = threadIdx.x; i < H; i += 256) {
        float v = xr[i] + (rr ? rr[i] : 0.f);
        float y = (v - mean) * rinv * g[i] + b[i];
        long long off = (long long)row * H + i;
        if (outf) outf[off] = y;
        if (outb) outb[off] = f2bf(y);
    }
}

// softmax over a row (scaled); writes bf16, zero-fills [valid, pad). grid = (rows, Z).
__global__ __launch_bounds__(256) void k_softmax(const float* __restrict__ S, long long ldS, long long sS,
                                                 bf16_t* __restrict__ P, long long ldP, long long sP,
                                                 int valid, int pad, float scale) {
    const float* sr = S + (long long)blockIdx.y * sS + (long long)blockIdx.x * ldS;
    bf16_t*      pr = P + (long long)blockIdx.y * sP + (long long)blockIdx.x * ldP;
    __shared__ float red[256];
    float m = -3.4e38f;
    for (int i = threadIdx.x; i < valid; i += 256) m = fmaxf(m, sr[i] * scale);
    red[threadIdx.x] = m;  __syncthreads();
    for (int o = 128; o > 0; o >>= 1) {
        if (threadIdx.x < o) red[threadIdx.x] = fmaxf(red[threadIdx.x], red[threadIdx.x + o]);
        __syncthreads();
    }
    float gm = red[0];
    __syncthreads();
    float s = 0.f;
    for (int i = threadIdx.x; i < valid; i += 256) s += expf(sr[i] * scale - gm);
    red[threadIdx.x] = s;  __syncthreads();
    for (int o = 128; o > 0; o >>= 1) {
        if (threadIdx.x < o) red[threadIdx.x] += red[threadIdx.x + o];
        __syncthreads();
    }
    float rinv = 1.0f / red[0];
    for (int i = threadIdx.x; i < pad; i += 256)
        pr[i] = (i < valid) ? f2bf(expf(sr[i] * scale - gm) * rinv) : f2bf(0.0f);
}

// split (B*Ls, 3H) -> per-head Q (Z,Lpad,HD), K (Z,Lpad,HD), V^T (Z,HD,Lpad); zero-pad l>=Ls
__global__ __launch_bounds__(256) void k_repack_qkv(const bf16_t* __restrict__ qkv,
                                                    bf16_t* __restrict__ qh,
                                                    bf16_t* __restrict__ kh,
                                                    bf16_t* __restrict__ vT,
                                                    int Ls, int Lpad) {
    long long idx = (long long)blockIdx.x * 256 + threadIdx.x;
    long long total = (long long)BB * NH * Lpad * HD;
    if (idx >= total) return;
    int d = (int)(idx % HD);
    int l = (int)((idx / HD) % Lpad);
    int z = (int)(idx / ((long long)HD * Lpad));
    int h = z % NH, b = z / NH;
    bf16_t q = f2bf(0.f), k = f2bf(0.f), v = f2bf(0.f);
    if (l < Ls) {
        long long base = (long long)(b * Ls + l) * (3 * HDIM);
        int col = h * HD + d;
        q = qkv[base + col];
        k = qkv[base + HDIM + col];
        v = qkv[base + 2 * HDIM + col];
    }
    qh[((long long)z * Lpad + l) * HD + d] = q;
    kh[((long long)z * Lpad + l) * HD + d] = k;
    vT[((long long)z * HD + d) * Lpad + l] = v;
}

// (B*Ls, H) -> (Z, Lpad, HD), zero-pad
__global__ __launch_bounds__(256) void k_repack_q(const bf16_t* __restrict__ src,
                                                  bf16_t* __restrict__ qh, int Ls, int Lpad) {
    long long idx = (long long)blockIdx.x * 256 + threadIdx.x;
    long long total = (long long)BB * NH * Lpad * HD;
    if (idx >= total) return;
    int d = (int)(idx % HD);
    int l = (int)((idx / HD) % Lpad);
    int z = (int)(idx / ((long long)HD * Lpad));
    int h = z % NH, b = z / NH;
    bf16_t q = f2bf(0.f);
    if (l < Ls) q = src[(long long)(b * Ls + l) * HDIM + h * HD + d];
    qh[((long long)z * Lpad + l) * HD + d] = q;
}

// cross K/V: (B*512, 2H) -> K (Z,512,HD), V^T (Z,HD,512)
__global__ __launch_bounds__(256) void k_repack_kv(const bf16_t* __restrict__ kv,
                                                   bf16_t* __restrict__ kh,
                                                   bf16_t* __restrict__ vT) {
    long long idx = (long long)blockIdx.x * 256 + threadIdx.x;
    long long total = (long long)BB * NH * LIN * HD;
    if (idx >= total) return;
    int d = (int)(idx % HD);
    int l = (int)((idx / HD) % LIN);
    int z = (int)(idx / ((long long)HD * LIN));
    int h = z % NH, b = z / NH;
    long long base = (long long)(b * LIN + l) * (2 * HDIM);
    kh[((long long)z * LIN + l) * HD + d] = kv[base + h * HD + d];
    vT[((long long)z * HD + d) * LIN + l] = kv[base + HDIM + h * HD + d];
}

// (Z, Lpad, HD) -> (B*Ls, H) merged heads
__global__ __launch_bounds__(256) void k_merge(const bf16_t* __restrict__ O,
                                               bf16_t* __restrict__ out, int Ls, int Lpad) {
    long long idx = (long long)blockIdx.x * 256 + threadIdx.x;
    long long total = (long long)BB * Ls * HDIM;
    if (idx >= total) return;
    int f = (int)(idx % HDIM);
    long long tok = idx / HDIM;
    int l = (int)(tok % Ls), b = (int)(tok / Ls);
    int h = f / HD, d = f % HD;
    out[idx] = O[(((long long)(b * NH + h)) * Lpad + l) * HD + d];
}

__global__ __launch_bounds__(256) void k_build_tin(const float* __restrict__ tcur,
                                                   bf16_t* __restrict__ tin, int Ls) {
    long long idx = (long long)blockIdx.x * 256 + threadIdx.x;
    long long total = (long long)BB * Ls * FOUT;
    if (idx >= total) return;
    int f = (int)(idx % FOUT);
    int l = (int)((idx / FOUT) % Ls);
    int b = (int)(idx / ((long long)FOUT * Ls));
    tin[idx] = f2bf(tcur[((long long)b * (OUTSEQ + 1) + l) * FOUT + f]);
}

__global__ __launch_bounds__(256) void k_append(const float* __restrict__ tfin,
                                                float* __restrict__ tcur,
                                                float* __restrict__ out, int Ls, int step) {
    int idx = blockIdx.x * 256 + threadIdx.x;
    if (idx >= BB * FOUT) return;
    int f = idx % FOUT, b = idx / FOUT;
    float v = tfin[((long long)(b * Ls + (Ls - 1))) * FOUT + f];
    tcur[((long long)b * (OUTSEQ + 1) + Ls) * FOUT + f] = v;
    out[((long long)b * OUTSEQ + step) * FOUT + f] = v;
}

// ---------------- host side ----------------
static inline void* bump(char*& p, size_t bytes) {
    void* r = (void*)p;
    p += (bytes + 255) & ~(size_t)255;
    return r;
}
static inline int cdiv(long long n, int d) { return (int)((n + d - 1) / d); }

extern "C" void kernel_launch(void* const* d_in, const int* in_sizes, int n_in,
                              void* d_out, int out_size, void* d_ws, size_t ws_size,
                              hipStream_t stream) {
    (void)in_sizes; (void)n_in; (void)out_size; (void)ws_size;
    // inputs in setup_inputs() dict insertion order (recursive over nested dicts)
    const float* X       = (const float*)d_in[0];   // (16,512,64)
    /* d_in[1] = t, unused by reference */
    const float* emb_w   = (const float*)d_in[2];
    const float* emb_b   = (const float*)d_in[3];
    const float* embt_w  = (const float*)d_in[4];
    const float* embt_b  = (const float*)d_in[5];
    const float* fc_w    = (const float*)d_in[6];
    const float* fc_b    = (const float*)d_in[7];
    const float* e_qkv_w = (const float*)d_in[8];
    const float* e_qkv_b = (const float*)d_in[9];
    const float* e_out_w = (const float*)d_in[10];
    const float* e_out_b = (const float*)d_in[11];
    const float* e_ff1_w = (const float*)d_in[12];
    const float* e_ff1_b = (const float*)d_in[13];
    const float* e_ff2_w = (const float*)d_in[14];
    const float* e_ff2_b = (const float*)d_in[15];
    const float* e_ln1_g = (const float*)d_in[16];
    const float* e_ln1_b = (const float*)d_in[17];
    const float* e_ln2_g = (const float*)d_in[18];
    const float* e_ln2_b = (const float*)d_in[19];
    const float* enc_ng  = (const float*)d_in[20];
    const float* enc_nb  = (const float*)d_in[21];
    const float* sa_qkv_w= (const float*)d_in[22];
    const float* sa_qkv_b= (const float*)d_in[23];
    const float* sa_out_w= (const float*)d_in[24];
    const float* sa_out_b= (const float*)d_in[25];
    const float* ca_qkv_w= (const float*)d_in[26];
    const float* ca_qkv_b= (const float*)d_in[27];
    const float* ca_out_w= (const float*)d_in[28];
    const float* ca_out_b= (const float*)d_in[29];
    const float* d_ff1_w = (const float*)d_in[30];
    const float* d_ff1_b = (const float*)d_in[31];
    const float* d_ff2_w = (const float*)d_in[32];
    const float* d_ff2_b = (const float*)d_in[33];
    const float* d_ln1_g = (const float*)d_in[34];
    const float* d_ln1_b = (const float*)d_in[35];
    const float* d_ln2_g = (const float*)d_in[36];
    const float* d_ln2_b = (const float*)d_in[37];
    const float* d_ln3_g = (const float*)d_in[38];
    const float* d_ln3_b = (const float*)d_in[39];
    const float* dec_ng  = (const float*)d_in[40];
    const float* dec_nb  = (const float*)d_in[41];
    float* OUT = (float*)d_out;

    // ------- workspace carve -------
    char* wp = (char*)d_ws;
    const int TOK = BB * LIN;                         // 8192
    bf16_t* w_emb  = (bf16_t*)bump(wp, (size_t)HDIM * FIN * 2);
    bf16_t* w_embt = (bf16_t*)bump(wp, (size_t)HDIM * FOUT * 2);
    bf16_t* w_fc   = (bf16_t*)bump(wp, (size_t)FOUT * HDIM * 2);
    bf16_t* w_eqkv = (bf16_t*)bump(wp, (size_t)NLAY * 3 * HDIM * HDIM * 2);
    bf16_t* w_eout = (bf16_t*)bump(wp, (size_t)NLAY * HDIM * HDIM * 2);
    bf16_t* w_eff1 = (bf16_t*)bump(wp, (size_t)NLAY * FFD * HDIM * 2);
    bf16_t* w_eff2 = (bf16_t*)bump(wp, (size_t)NLAY * HDIM * FFD * 2);
    bf16_t* w_saqkv= (bf16_t*)bump(wp, (size_t)NLAY * 3 * HDIM * HDIM * 2);
    bf16_t* w_saout= (bf16_t*)bump(wp, (size_t)NLAY * HDIM * HDIM * 2);
    bf16_t* w_caqkv= (bf16_t*)bump(wp, (size_t)NLAY * 3 * HDIM * HDIM * 2);
    bf16_t* w_caout= (bf16_t*)bump(wp, (size_t)NLAY * HDIM * HDIM * 2);
    bf16_t* w_dff1 = (bf16_t*)bump(wp, (size_t)NLAY * FFD * HDIM * 2);
    bf16_t* w_dff2 = (bf16_t*)bump(wp, (size_t)NLAY * HDIM * FFD * 2);
    bf16_t* x_bf   = (bf16_t*)bump(wp, (size_t)TOK * FIN * 2);
    float*  hbuf   = (float*) bump(wp, (size_t)TOK * HDIM * 4);
    bf16_t* h_bf   = (bf16_t*)bump(wp, (size_t)TOK * HDIM * 2);
    float*  t0     = (float*) bump(wp, (size_t)TOK * HDIM * 4);
    char*   big    = (char*)  bump(wp, (size_t)TOK * FFD * 2);   // 32 MiB multi-use
    bf16_t* qh     = (bf16_t*)bump(wp, (size_t)BB * NH * LIN * HD * 2);
    bf16_t* kh     = (bf16_t*)bump(wp, (size_t)BB * NH * LIN * HD * 2);
    bf16_t* vT     = (bf16_t*)bump(wp, (size_t)BB * NH * LIN * HD * 2);
    bf16_t* Obuf   = (bf16_t*)bump(wp, (size_t)BB * NH * LIN * HD * 2);
    bf16_t* attnA  = (bf16_t*)bump(wp, (size_t)TOK * HDIM * 2);
    float*  sc     = (float*) bump(wp, (size_t)NH * LIN * LIN * 4);     // 8 MiB
    bf16_t* pbf    = (bf16_t*)bump(wp, (size_t)NH * LIN * LIN * 2);     // 4 MiB
    bf16_t* mem_bf = (bf16_t*)bump(wp, (size_t)TOK * HDIM * 2);
    bf16_t* sqh    = (bf16_t*)bump(wp, (size_t)BB * NH * 32 * HD * 2);
    bf16_t* skh    = (bf16_t*)bump(wp, (size_t)BB * NH * 32 * HD * 2);
    bf16_t* svT    = (bf16_t*)bump(wp, (size_t)BB * NH * HD * 32 * 2);
    bf16_t* cqh    = (bf16_t*)bump(wp, (size_t)BB * NH * 32 * HD * 2);
    bf16_t* decn   = (bf16_t*)bump(wp, (size_t)BB * OUTSEQ * HDIM * 2);
    bf16_t* tin    = (bf16_t*)bump(wp, (size_t)BB * OUTSEQ * FOUT * 2);
    float*  tfin   = (float*) bump(wp, (size_t)BB * OUTSEQ * FOUT * 4);
    float*  tcur   = (float*) bump(wp, (size_t)BB * (OUTSEQ + 1) * FOUT * 4);
    // aliases inside `big` (disjoint in time or offset)
    bf16_t* enc_qkv_bf = (bf16_t*)big;                             // (8192,1536)
    bf16_t* enc_ff1_bf = (bf16_t*)big;                             // (8192,2048)
    bf16_t* dqkv_bf    = (bf16_t*)big;                             // (M,1536)
    bf16_t* ckv_bf     = (bf16_t*)big;                             // (8192,1024)
    bf16_t* cq_bf      = (bf16_t*)(big + (size_t)24 * 1024 * 1024);
    bf16_t* dff1_bf    = (bf16_t*)(big + (size_t)28 * 1024 * 1024);

    auto cvt = [&](const float* s, bf16_t* d, long long n) {
        k_cvt<<<cdiv(n, 256), 256, 0, stream>>>(s, d, n);
    };
    auto gemm = [&](const bf16_t* A, long long lda, long long sA,
                    const bf16_t* W, long long ldw, long long sW,
                    const float* bias, float* Cf, long long sCf,
                    bf16_t* Cb, long long sCb, long long ldc,
                    int M, int N, int K, int batch, int relu) {
        dim3 g((N + 63) / 64, (M + 127) / 128, batch);
        k_gemm<<<g, 256, 0, stream>>>(A, lda, sA, W, ldw, sW, bias, Cf, sCf, Cb, sCb, ldc, M, N, K, relu);
    };

    // ------- convert all weights + input to bf16 (once per call; deterministic) -------
    cvt(emb_w,  w_emb,  (long long)HDIM * FIN);
    cvt(embt_w, w_embt, (long long)HDIM * FOUT);
    cvt(fc_w,   w_fc,   (long long)FOUT * HDIM);
    cvt(e_qkv_w, w_eqkv, (long long)NLAY * 3 * HDIM * HDIM);
    cvt(e_out_w, w_eout, (long long)NLAY * HDIM * HDIM);
    cvt(e_ff1_w, w_eff1, (long long)NLAY * FFD * HDIM);
    cvt(e_ff2_w, w_eff2, (long long)NLAY * HDIM * FFD);
    cvt(sa_qkv_w, w_saqkv, (long long)NLAY * 3 * HDIM * HDIM);
    cvt(sa_out_w, w_saout, (long long)NLAY * HDIM * HDIM);
    cvt(ca_qkv_w, w_caqkv, (long long)NLAY * 3 * HDIM * HDIM);
    cvt(ca_out_w, w_caout, (long long)NLAY * HDIM * HDIM);
    cvt(d_ff1_w, w_dff1, (long long)NLAY * FFD * HDIM);
    cvt(d_ff2_w, w_dff2, (long long)NLAY * HDIM * FFD);
    cvt(X, x_bf, (long long)TOK * FIN);
    k_zero<<<cdiv((long long)BB * (OUTSEQ + 1) * FOUT, 256), 256, 0, stream>>>(
        tcur, (long long)BB * (OUTSEQ + 1) * FOUT);

    // =================== ENCODER ===================
    gemm(x_bf, FIN, 0, w_emb, FIN, 0, emb_b, t0, 0, nullptr, 0, HDIM, TOK, HDIM, FIN, 1, 0);
    k_rope<<<TOK, 256, 0, stream>>>(t0, hbuf, h_bf, LIN);

    for (int L = 0; L < NLAY; ++L) {
        const bf16_t* wq = w_eqkv + (size_t)L * 3 * HDIM * HDIM;
        const bf16_t* wo = w_eout + (size_t)L * HDIM * HDIM;
        const bf16_t* w1 = w_eff1 + (size_t)L * FFD * HDIM;
        const bf16_t* w2 = w_eff2 + (size_t)L * HDIM * FFD;
        // QKV projection
        gemm(h_bf, HDIM, 0, wq, HDIM, 0, e_qkv_b + L * 3 * HDIM,
             nullptr, 0, enc_qkv_bf, 0, 3 * HDIM, TOK, 3 * HDIM, HDIM, 1, 0);
        k_repack_qkv<<<cdiv((long long)BB * NH * LIN * HD, 256), 256, 0, stream>>>(
            enc_qkv_bf, qh, kh, vT, LIN, LIN);
        // attention, chunked per batch element (8 heads)
        for (int b = 0; b < BB; ++b) {
            long long z0 = (long long)b * NH;
            gemm(qh + z0 * LIN * HD, HD, (long long)LIN * HD,
                 kh + z0 * LIN * HD, HD, (long long)LIN * HD,
                 nullptr, sc, (long long)LIN * LIN, nullptr, 0, LIN,
                 LIN, LIN, HD, NH, 0);
            k_softmax<<<dim3(LIN, NH), 256, 0, stream>>>(
                sc, LIN, (long long)LIN * LIN, pbf, LIN, (long long)LIN * LIN,
                LIN, LIN, 0.125f);
            gemm(pbf, LIN, (long long)LIN * LIN,
                 vT + z0 * HD * LIN, LIN, (long long)HD * LIN,
                 nullptr, nullptr, 0, Obuf + z0 * LIN * HD, (long long)LIN * HD, HD,
                 LIN, HD, LIN, NH, 0);
        }
        k_merge<<<cdiv((long long)TOK * HDIM, 256), 256, 0, stream>>>(Obuf, attnA, LIN, LIN);
        gemm(attnA, HDIM, 0, wo, HDIM, 0, e_out_b + L * HDIM,
             t0, 0, nullptr, 0, HDIM, TOK, HDIM, HDIM, 1, 0);
        k_ln<<<TOK, 256, 0, stream>>>(t0, hbuf, e_ln1_g + L * HDIM, e_ln1_b + L * HDIM, hbuf, h_bf, HDIM);
        // FFN
        gemm(h_bf, HDIM, 0, w1, HDIM, 0, e_ff1_b + L * FFD,
             nullptr, 0, enc_ff1_bf, 0, FFD, TOK, FFD, HDIM, 1, 1);
        gemm(enc_ff1_bf, FFD, 0, w2, FFD, 0, e_ff2_b + L * HDIM,
             t0, 0, nullptr, 0, HDIM, TOK, HDIM, FFD, 1, 0);
        k_ln<<<TOK, 256, 0, stream>>>(t0, hbuf, e_ln2_g + L * HDIM, e_ln2_b + L * HDIM, hbuf, h_bf, HDIM);
    }
    k_ln<<<TOK, 256, 0, stream>>>(hbuf, nullptr, enc_ng, enc_nb, nullptr, mem_bf, HDIM);

    // =================== DECODER ===================
    for (int step = 0; step < OUTSEQ; ++step) {
        const int Ls = step + 1;
        const int M  = BB * Ls;
        k_build_tin<<<cdiv((long long)M * FOUT, 256), 256, 0, stream>>>(tcur, tin, Ls);
        gemm(tin, FOUT, 0, w_embt, FOUT, 0, embt_b, t0, 0, nullptr, 0, HDIM, M, HDIM, FOUT, 1, 0);
        k_rope<<<M, 256, 0, stream>>>(t0, hbuf, h_bf, Ls);

        for (int L = 0; L < NLAY; ++L) {
            const bf16_t* wsq = w_saqkv + (size_t)L * 3 * HDIM * HDIM;
            const bf16_t* wso = w_saout + (size_t)L * HDIM * HDIM;
            const bf16_t* wcq = w_caqkv + (size_t)L * 3 * HDIM * HDIM;
            const bf16_t* wco = w_caout + (size_t)L * HDIM * HDIM;
            const bf16_t* w1  = w_dff1  + (size_t)L * FFD * HDIM;
            const bf16_t* w2  = w_dff2  + (size_t)L * HDIM * FFD;
            // ---- self attention (no mask; full recompute each step, K padded to 32) ----
            gemm(h_bf, HDIM, 0, wsq, HDIM, 0, sa_qkv_b + L * 3 * HDIM,
                 nullptr, 0, dqkv_bf, 0, 3 * HDIM, M, 3 * HDIM, HDIM, 1, 0);
            k_repack_qkv<<<cdiv((long long)BB * NH * 32 * HD, 256), 256, 0, stream>>>(
                dqkv_bf, sqh, skh, svT, Ls, 32);
            gemm(sqh, HD, 32 * HD, skh, HD, 32 * HD, nullptr,
                 sc, 32 * 32, nullptr, 0, 32, Ls, Ls, HD, BB * NH, 0);
            k_softmax<<<dim3(Ls, BB * NH), 256, 0, stream>>>(
                sc, 32, 32 * 32, pbf, 32, 32 * 32, Ls, 32, 0.125f);
            gemm(pbf, 32, 32 * 32, svT, 32, HD * 32, nullptr,
                 nullptr, 0, Obuf, 32 * HD, HD, Ls, HD, 32, BB * NH, 0);
            k_merge<<<cdiv((long long)M * HDIM, 256), 256, 0, stream>>>(Obuf, attnA, Ls, 32);
            gemm(attnA, HDIM, 0, wso, HDIM, 0, sa_out_b + L * HDIM,
                 t0, 0, nullptr, 0, HDIM, M, HDIM, HDIM, 1, 0);
            k_ln<<<M, 256, 0, stream>>>(t0, hbuf, d_ln1_g + L * HDIM, d_ln1_b + L * HDIM, hbuf, h_bf, HDIM);
            // ---- cross attention ----
            gemm(h_bf, HDIM, 0, wcq, HDIM, 0, ca_qkv_b + L * 3 * HDIM,
                 nullptr, 0, cq_bf, 0, HDIM, M, HDIM, HDIM, 1, 0);
            gemm(mem_bf, HDIM, 0, wcq + (size_t)HDIM * HDIM, HDIM, 0, ca_qkv_b + L * 3 * HDIM + HDIM,
                 nullptr, 0, ckv_bf, 0, 2 * HDIM, TOK, 2 * HDIM, HDIM, 1, 0);
            k_repack_q<<<cdiv((long long)BB * NH * 32 * HD, 256), 256, 0, stream>>>(cq_bf, cqh, Ls, 32);
            k_repack_kv<<<cdiv((long long)BB * NH * LIN * HD, 256), 256, 0, stream>>>(ckv_bf, kh, vT);
            gemm(cqh, HD, 32 * HD, kh, HD, (long long)LIN * HD, nullptr,
                 sc, 32 * LIN, nullptr, 0, LIN, Ls, LIN, HD, BB * NH, 0);
            k_softmax<<<dim3(Ls, BB * NH), 256, 0, stream>>>(
                sc, LIN, 32 * LIN, pbf, LIN, 32 * LIN, LIN, LIN, 0.125f);
            gemm(pbf, LIN, 32 * LIN, vT, LIN, (long long)HD * LIN, nullptr,
                 nullptr, 0, Obuf, 32 * HD, HD, Ls, HD, LIN, BB * NH, 0);
            k_merge<<<cdiv((long long)M * HDIM, 256), 256, 0, stream>>>(Obuf, attnA, Ls, 32);
            gemm(attnA, HDIM, 0, wco, HDIM, 0, ca_out_b + L * HDIM,
                 t0, 0, nullptr, 0, HDIM, M, HDIM, HDIM, 1, 0);
            k_ln<<<M, 256, 0, stream>>>(t0, hbuf, d_ln2_g + L * HDIM, d_ln2_b + L * HDIM, hbuf, h_bf, HDIM);
            // ---- FFN ----
            gemm(h_bf, HDIM, 0, w1, HDIM, 0, d_ff1_b + L * FFD,
                 nullptr, 0, dff1_bf, 0, FFD, M, FFD, HDIM, 1, 1);
            gemm(dff1_bf, FFD, 0, w2, FFD, 0, d_ff2_b + L * HDIM,
                 t0, 0, nullptr, 0, HDIM, M, HDIM, FFD, 1, 0);
            k_ln<<<M, 256, 0, stream>>>(t0, hbuf, d_ln3_g + L * HDIM, d_ln3_b + L * HDIM, hbuf, h_bf, HDIM);
        }
        k_ln<<<M, 256, 0, stream>>>(hbuf, nullptr, dec_ng, dec_nb, nullptr, decn, HDIM);
        gemm(decn, HDIM, 0, w_fc, HDIM, 0, fc_b, tfin, 0, nullptr, 0, FOUT, M, FOUT, HDIM, 1, 0);
        k_append<<<cdiv(BB * FOUT, 256), 256, 0, stream>>>(tfin, tcur, OUT, Ls, step);
    }
}
